// ExpertRouter_34806414967252
// MI455X (gfx1250) — matrix operand order repaired and checked by reading, not compiled, not run on hardware
//
#include <hip/hip_runtime.h>
#include <hip/hip_bf16.h>

// ---------------------------------------------------------------------------
// MoE Expert Router for MI455X (gfx1250, wave32, WMMA).
// logits = hs[16384,4096] x gate_w^T[4096,64] via v_wmma_f32_16x16x32_bf16,
// software-pipelined (double-buffered A/B fragments), 2 M-tiles per wave,
// fused top-2 / softmax / Switch aux-loss accumulation.
// ---------------------------------------------------------------------------

typedef __attribute__((ext_vector_type(16))) __bf16 v16bf;
typedef __attribute__((ext_vector_type(8)))  float  v8f;

#define NUM_TOKENS   16384          // B*S = 4*4096
#define HDIM         4096
#define NEXPERT      64
#define MTILE        128            // tokens per workgroup (4 waves x 32 rows)
#define BLOCK        128            // 4 waves
#define LSTRIDE      65             // LDS logits row stride (odd -> no bank conflicts)
#define ALPHA        0.01f

#define WMMA_BF16(a, b, c) \
    __builtin_amdgcn_wmma_f32_16x16x32_bf16(false, (a), false, (b), (short)0, (c), false, false)

static __device__ __forceinline__ float4 ld4(const float* p) {
    return *reinterpret_cast<const float4*>(p);
}

// Build bf16 A fragment (contiguous K order: element s is k = 16*laneHi + s) from 16 fp32 values:
// one contiguous 16-element K-chunk per lane.
static __device__ __forceinline__ v16bf cvt_a(float4 q0, float4 q1, float4 q2, float4 q3) {
    v16bf a;
    a[0]  = (__bf16)q0.x; a[1]  = (__bf16)q0.y; a[2]  = (__bf16)q0.z; a[3]  = (__bf16)q0.w;
    a[4]  = (__bf16)q1.x; a[5]  = (__bf16)q1.y; a[6]  = (__bf16)q1.z; a[7]  = (__bf16)q1.w;
    a[8]  = (__bf16)q2.x; a[9]  = (__bf16)q2.y; a[10] = (__bf16)q2.z; a[11] = (__bf16)q2.w;
    a[12] = (__bf16)q3.x; a[13] = (__bf16)q3.y; a[14] = (__bf16)q3.z; a[15] = (__bf16)q3.w;
    return a;
}

// ---------------- kernel 0: gate fp32 -> bf16, zero accumulators -----------
__global__ __launch_bounds__(256)
void router_prep(const float* __restrict__ gate_w,
                 __hip_bfloat16* __restrict__ gate_bf,
                 float* __restrict__ p_acc, float* __restrict__ c_acc) {
    int i = blockIdx.x * blockDim.x + threadIdx.x;
    if (i < NEXPERT * HDIM) gate_bf[i] = __float2bfloat16(gate_w[i]);
    if (blockIdx.x == 0 && threadIdx.x < NEXPERT) {
        p_acc[threadIdx.x] = 0.0f;
        c_acc[threadIdx.x] = 0.0f;
    }
}

// ---------------- kernel 1: fused GEMM + routing ---------------------------
__global__ __launch_bounds__(BLOCK)
void router_main(const float* __restrict__ hidden,
                 const __hip_bfloat16* __restrict__ gate_bf,
                 float* __restrict__ out_w,      // [NUM_TOKENS*2]
                 float* __restrict__ out_sel,    // [NUM_TOKENS*2] (indices as float)
                 float* __restrict__ p_acc,      // [64] sum of router_probs
                 float* __restrict__ c_acc) {    // [64] tokens routed (top-2 hits)
    __shared__ float lg[MTILE * LSTRIDE];
    __shared__ float lmax[MTILE];
    __shared__ float linvs[MTILE];
    __shared__ int   li0[MTILE];
    __shared__ int   li1[MTILE];

    const int tid    = threadIdx.x;
    const int wave   = tid >> 5;
    const int lane   = tid & 31;
    const int laneLo = lane & 15;
    const int laneHi = lane >> 4;          // 0 or 1
    const int wgBase = blockIdx.x * MTILE;

    // Each wave: 32 tokens (2 M-tiles of 16) x all 64 experts.
    // A fragment: lane holds row laneLo; 16 consecutive K at
    // k0+16*laneHi.
    const int aRow0 = wgBase + wave * 32 + laneLo;
    const float* h0 = hidden + (size_t)aRow0 * HDIM + 16 * laneHi;
    const float* h1 = h0 + (size_t)16 * HDIM;           // second M-tile

    // B fragment: lane holds expert column eb*16+laneLo, 16 consecutive K at
    // k0+16*laneHi -> contiguous in gate_bf [E,H] (row-major = N-major).
    const __bf16* gb = reinterpret_cast<const __bf16*>(gate_bf) + 16 * laneHi;
    const __bf16* g0 = gb + (size_t)(0  + laneLo) * HDIM;
    const __bf16* g1 = gb + (size_t)(16 + laneLo) * HDIM;
    const __bf16* g2 = gb + (size_t)(32 + laneLo) * HDIM;
    const __bf16* g3 = gb + (size_t)(48 + laneLo) * HDIM;

    v8f c00 = {}, c01 = {}, c02 = {}, c03 = {};   // M-tile 0, expert blocks 0..3
    v8f c10 = {}, c11 = {}, c12 = {}, c13 = {};   // M-tile 1

    // ---- software-pipeline prologue: fetch k0 = 0 fragments ----
    float4 qa00 = ld4(h0 + 0), qa01 = ld4(h0 + 4), qa02 = ld4(h0 + 8), qa03 = ld4(h0 + 12);
    float4 qa10 = ld4(h1 + 0), qa11 = ld4(h1 + 4), qa12 = ld4(h1 + 8), qa13 = ld4(h1 + 12);
    v16bf bb0 = *reinterpret_cast<const v16bf*>(g0);
    v16bf bb1 = *reinterpret_cast<const v16bf*>(g1);
    v16bf bb2 = *reinterpret_cast<const v16bf*>(g2);
    v16bf bb3 = *reinterpret_cast<const v16bf*>(g3);

    // ---- steady state: issue k0+32 loads, then consume k0 fragments ----
    for (int k0 = 0; k0 < HDIM - 32; k0 += 32) {
        const int kn = k0 + 32;
        const float4 na00 = ld4(h0 + kn),      na01 = ld4(h0 + kn + 4);
        const float4 na02 = ld4(h0 + kn + 8), na03 = ld4(h0 + kn + 12);
        const float4 na10 = ld4(h1 + kn),      na11 = ld4(h1 + kn + 4);
        const float4 na12 = ld4(h1 + kn + 8), na13 = ld4(h1 + kn + 12);
        const v16bf nb0 = *reinterpret_cast<const v16bf*>(g0 + kn);
        const v16bf nb1 = *reinterpret_cast<const v16bf*>(g1 + kn);
        const v16bf nb2 = *reinterpret_cast<const v16bf*>(g2 + kn);
        const v16bf nb3 = *reinterpret_cast<const v16bf*>(g3 + kn);
        __builtin_prefetch(h0 + kn + 256, 0, 3);   // global_prefetch_b8, 8 steps ahead
        __builtin_prefetch(h1 + kn + 256, 0, 3);

        const v16bf a0 = cvt_a(qa00, qa01, qa02, qa03);
        const v16bf a1 = cvt_a(qa10, qa11, qa12, qa13);
        c00 = WMMA_BF16(a0, bb0, c00);
        c01 = WMMA_BF16(a0, bb1, c01);
        c02 = WMMA_BF16(a0, bb2, c02);
        c03 = WMMA_BF16(a0, bb3, c03);
        c10 = WMMA_BF16(a1, bb0, c10);
        c11 = WMMA_BF16(a1, bb1, c11);
        c12 = WMMA_BF16(a1, bb2, c12);
        c13 = WMMA_BF16(a1, bb3, c13);

        qa00 = na00; qa01 = na01; qa02 = na02; qa03 = na03;
        qa10 = na10; qa11 = na11; qa12 = na12; qa13 = na13;
        bb0 = nb0; bb1 = nb1; bb2 = nb2; bb3 = nb3;
    }
    // ---- epilogue: last K-step ----
    {
        const v16bf a0 = cvt_a(qa00, qa01, qa02, qa03);
        const v16bf a1 = cvt_a(qa10, qa11, qa12, qa13);
        c00 = WMMA_BF16(a0, bb0, c00);
        c01 = WMMA_BF16(a0, bb1, c01);
        c02 = WMMA_BF16(a0, bb2, c02);
        c03 = WMMA_BF16(a0, bb3, c03);
        c10 = WMMA_BF16(a1, bb0, c10);
        c11 = WMMA_BF16(a1, bb1, c11);
        c12 = WMMA_BF16(a1, bb2, c12);
        c13 = WMMA_BF16(a1, bb3, c13);
    }

    // C/D layout: VGPR r -> row r (lanes 0-15) or 8+r (lanes 16-31), col = laneLo
#pragma unroll
    for (int r = 0; r < 8; ++r) {
        const int row0 = wave * 32 + r + 8 * laneHi;        // M-tile 0
        const int row1 = row0 + 16;                          // M-tile 1
        lg[row0 * LSTRIDE +  0 + laneLo] = c00[r];
        lg[row0 * LSTRIDE + 16 + laneLo] = c01[r];
        lg[row0 * LSTRIDE + 32 + laneLo] = c02[r];
        lg[row0 * LSTRIDE + 48 + laneLo] = c03[r];
        lg[row1 * LSTRIDE +  0 + laneLo] = c10[r];
        lg[row1 * LSTRIDE + 16 + laneLo] = c11[r];
        lg[row1 * LSTRIDE + 32 + laneLo] = c12[r];
        lg[row1 * LSTRIDE + 48 + laneLo] = c13[r];
    }
    __syncthreads();

    // --- phase 1: one thread per token: top-2, softmax(top-2), outputs -----
    {
        const int base = tid * LSTRIDE;
        float v0 = -3.402823466e+38f, v1 = -3.402823466e+38f;
        int   i0 = 0, i1 = 0;
#pragma unroll 8
        for (int e = 0; e < NEXPERT; ++e) {
            const float l = lg[base + e];
            if (l > v0)      { v1 = v0; i1 = i0; v0 = l; i0 = e; }
            else if (l > v1) { v1 = l;  i1 = e; }
        }
        float s = 0.0f;
#pragma unroll 8
        for (int e = 0; e < NEXPERT; ++e) s += __expf(lg[base + e] - v0);

        const float e1  = __expf(v1 - v0);         // softmax over {v0, v1}, v0 >= v1
        const float inv = 1.0f / (1.0f + e1);
        const int t = wgBase + tid;
        out_w[t * 2 + 0]   = inv;
        out_w[t * 2 + 1]   = e1 * inv;
        out_sel[t * 2 + 0] = (float)i0;
        out_sel[t * 2 + 1] = (float)i1;

        lmax[tid]  = v0;
        linvs[tid] = 1.0f / s;
        li0[tid]   = i0;
        li1[tid]   = i1;
    }
    __syncthreads();

    // --- phase 2: one thread per expert: P_i partial sum + top-2 counts ----
    if (tid < NEXPERT) {
        float accP = 0.0f, cnt = 0.0f;
#pragma unroll 4
        for (int t = 0; t < MTILE; ++t) {
            accP += __expf(lg[t * LSTRIDE + tid] - lmax[t]) * linvs[t];
            cnt  += (float)((li0[t] == tid) + (li1[t] == tid));
        }
        atomicAdd(&p_acc[tid], accP);
        atomicAdd(&c_acc[tid], cnt);
    }
}

// ---------------- kernel 2: finalize scalar aux loss -----------------------
__global__ __launch_bounds__(64)
void router_finalize(const float* __restrict__ p_acc,
                     const float* __restrict__ c_acc,
                     float* __restrict__ out_loss) {
    __shared__ float red[NEXPERT];
    const int e = threadIdx.x;
    red[e] = p_acc[e] * c_acc[e];
    __syncthreads();
    if (e == 0) {
        float s = 0.0f;
        for (int i = 0; i < NEXPERT; ++i) s += red[i];
        const float T = (float)NUM_TOKENS;
        *out_loss = ALPHA * (float)NEXPERT * s / (T * T);
    }
}

// ---------------------------------------------------------------------------
extern "C" void kernel_launch(void* const* d_in, const int* in_sizes, int n_in,
                              void* d_out, int out_size, void* d_ws, size_t ws_size,
                              hipStream_t stream) {
    const float* hidden = (const float*)d_in[0];   // [4,4096,4096] f32
    const float* gate_w = (const float*)d_in[1];   // [64,4096]     f32

    float* out      = (float*)d_out;
    float* out_w    = out;                          // [16384*2] router weights
    float* out_sel  = out + NUM_TOKENS * 2;         // [16384*2] indices (as float)
    float* out_loss = out + NUM_TOKENS * 4;         // [1] aux loss

    // workspace: bf16 gate (512 KB) | P_acc[64] | C_acc[64]
    __hip_bfloat16* gate_bf = (__hip_bfloat16*)d_ws;
    float* p_acc = (float*)((char*)d_ws + (size_t)NEXPERT * HDIM * 2);
    float* c_acc = p_acc + NEXPERT;

    router_prep<<<(NEXPERT * HDIM + 255) / 256, 256, 0, stream>>>(gate_w, gate_bf, p_acc, c_acc);
    router_main<<<NUM_TOKENS / MTILE, BLOCK, 0, stream>>>(hidden, gate_bf, out_w, out_sel, p_acc, c_acc);
    router_finalize<<<1, 64, 0, stream>>>(p_acc, c_acc, out_loss);
}
